// FeatureFusion_21861383537209
// MI455X (gfx1250) — matrix operand order repaired and checked by reading, not compiled, not run on hardware
//
#include <hip/hip_runtime.h>
#include <hip/hip_bf16.h>
#include <hip/hip_fp16.h>

// FeatureFusion for MI455X (gfx1250, wave32, WMMA).
//   G1: S = sigmoid(depth @ rgb^T)   (b,512,512)   staged in d_ws (fp32)
//   G2: out = rgb + S @ depth        (b,512,4096)
// bf16 inputs to v_wmma_f32_16x16x32_bf16, fp32 accumulate.

typedef __bf16 bf16_t;
typedef __attribute__((ext_vector_type(4)))  __bf16 v4bf;
typedef __attribute__((ext_vector_type(8)))  __bf16 v8bf;
typedef __attribute__((ext_vector_type(16))) __bf16 v16bf;
typedef __attribute__((ext_vector_type(8)))  float  v8f;

#define BATCH 16
#define CDIM  512
#define HWDIM 4096

#define BM 128
#define BN 128
#define BK 32
#define LDSS 40          // BK + 8 pad (bf16 elems); row stride 80 B spreads banks
#define NTHREADS 256

// ---- WMMA fragment loads from LDS (contiguous k order k = 16h + s, same for A and B) -----------
// A (16x32 MxK): lanes 0-15 row M=lane, K=0..15; lanes 16-31 same
// rows, K=16..31. Per-lane v16bf = [kh..kh+15].
__device__ __forceinline__ v16bf frag_a(const bf16_t* tile, int lane) {
  const int row = lane & 15;
  const int kh  = (lane >> 4) << 4;           // 0 or 16
  const bf16_t* p = tile + row * LDSS;
  v8bf lo = *(const v8bf*)(p + kh);
  v8bf hi = *(const v8bf*)(p + 8 + kh);
  return __builtin_shufflevector(lo, hi, 0,1,2,3,4,5,6,7,8,9,10,11,12,13,14,15);
}
// B (32x16 KxN): lanes 0-15 hold col N=lane, K=0..15; lanes 16-31 K=16..31.
// tile is stored [n][k] in LDS, so this is one contiguous 32-byte read.
__device__ __forceinline__ v16bf frag_b(const bf16_t* tile, int lane) {
  const int col = lane & 15;
  const int k   = (lane >> 4) << 4;           // 0 or 16
  return *(const v16bf*)(tile + col * LDSS + k);
}

// ============================================================================
// Kernel 1: S[b] = sigmoid( depth[b] (512x4096) @ rgb[b]^T )  -> ws (fp32)
// Both operands are K-contiguous rows in memory (K = hw).
// ============================================================================
__global__ __launch_bounds__(NTHREADS)
void ff_gemm1_sigmoid(const float* __restrict__ rgb,
                      const float* __restrict__ depth,
                      float* __restrict__ interaction) {
  __shared__ bf16_t sA[2][BM * LDSS];
  __shared__ bf16_t sB[2][BN * LDSS];

  const int t    = threadIdx.x;
  const int lane = t & 31;
  const int wave = t >> 5;        // 8 waves
  const int wm   = wave & 1;      // 2 wave-rows  x 64 rows
  const int wn   = wave >> 1;     // 4 wave-cols  x 32 cols

  const int batch = blockIdx.z;
  const int mbase = blockIdx.y * BM;
  const int nbase = blockIdx.x * BN;

  const float* Ag = depth + (size_t)batch * CDIM * HWDIM;  // rows = m, K contig
  const float* Bg = rgb   + (size_t)batch * CDIM * HWDIM;  // rows = n, K contig

  float4 ra[4], rb[4];

  auto gload = [&](int k0) {
#pragma unroll
    for (int p = 0; p < 4; ++p) {
      int idx = t + p * NTHREADS;          // 0..1023 float4 slots
      int r   = idx >> 3;                  // 0..127 tile row
      int kc  = (idx & 7) << 2;            // 0..28
      ra[p] = *(const float4*)(Ag + (size_t)(mbase + r) * HWDIM + k0 + kc);
      rb[p] = *(const float4*)(Bg + (size_t)(nbase + r) * HWDIM + k0 + kc);
    }
  };
  auto lstore = [&](int buf) {
#pragma unroll
    for (int p = 0; p < 4; ++p) {
      int idx = t + p * NTHREADS;
      int r   = idx >> 3;
      int kc  = (idx & 7) << 2;
      v4bf pa = { (bf16_t)ra[p].x, (bf16_t)ra[p].y, (bf16_t)ra[p].z, (bf16_t)ra[p].w };
      v4bf pb = { (bf16_t)rb[p].x, (bf16_t)rb[p].y, (bf16_t)rb[p].z, (bf16_t)rb[p].w };
      *(v4bf*)&sA[buf][r * LDSS + kc] = pa;
      *(v4bf*)&sB[buf][r * LDSS + kc] = pb;
    }
  };

  v8f acc[4][2] = {};

  gload(0);
  lstore(0);
  __syncthreads();

  const int NK = HWDIM / BK;               // 128 K-steps
  for (int kt = 0; kt < NK; ++kt) {
    const int cur = kt & 1;
    if (kt + 1 < NK) gload((kt + 1) * BK); // prefetch to regs, overlap compute

    v16bf af[4], bf[2];
#pragma unroll
    for (int i = 0; i < 4; ++i)
      af[i] = frag_a(&sA[cur][(wm * 64 + i * 16) * LDSS], lane);
#pragma unroll
    for (int j = 0; j < 2; ++j)
      bf[j] = frag_b(&sB[cur][(wn * 32 + j * 16) * LDSS], lane);

#pragma unroll
    for (int i = 0; i < 4; ++i)
#pragma unroll
      for (int j = 0; j < 2; ++j)
        acc[i][j] = __builtin_amdgcn_wmma_f32_16x16x32_bf16(
            false, af[i], false, bf[j], (short)0, acc[i][j], false, false);

    if (kt + 1 < NK) lstore(cur ^ 1);
    __syncthreads();
  }

  // epilogue: sigmoid -> ws.  C/D layout: VGPR v -> M = v + 8*(lane>=16), N = lane&15
  float* S = interaction + (size_t)batch * CDIM * CDIM;
  const int half = lane >> 4;
  const int ln   = lane & 15;
#pragma unroll
  for (int i = 0; i < 4; ++i) {
#pragma unroll
    for (int j = 0; j < 2; ++j) {
      const int n = nbase + wn * 32 + j * 16 + ln;
#pragma unroll
      for (int v = 0; v < 8; ++v) {
        const int m = mbase + wm * 64 + i * 16 + half * 8 + v;
        const float x = acc[i][j][v];
        S[(size_t)m * CDIM + n] = 1.0f / (1.0f + __expf(-x));
      }
    }
  }
}

// ============================================================================
// Kernel 2: out[b] = rgb[b] + S[b] (512x512) @ depth[b] (512x4096)
// A = S (K-contiguous rows). B = depth, row-major along N -> transpose into
// LDS [n][k] at store time so frag_b reads are contiguous.
// ============================================================================
__global__ __launch_bounds__(NTHREADS)
void ff_gemm2_add(const float* __restrict__ rgb,
                  const float* __restrict__ depth,
                  const float* __restrict__ interaction,
                  float* __restrict__ out) {
  __shared__ bf16_t sA[2][BM * LDSS];
  __shared__ bf16_t sB[2][BN * LDSS];

  const int t    = threadIdx.x;
  const int lane = t & 31;
  const int wave = t >> 5;
  const int wm   = wave & 1;
  const int wn   = wave >> 1;

  const int batch = blockIdx.z;
  const int mbase = blockIdx.y * BM;      // over C (512) -> 4
  const int nbase = blockIdx.x * BN;      // over HW (4096) -> 32

  const float* Ag = interaction + (size_t)batch * CDIM * CDIM;   // [m][k]
  const float* Dg = depth       + (size_t)batch * CDIM * HWDIM;  // [k][n]
  const float* Rg = rgb         + (size_t)batch * CDIM * HWDIM;
  float*       Og = out         + (size_t)batch * CDIM * HWDIM;

  float4 ra[4], rb[4];

  auto gload = [&](int k0) {
#pragma unroll
    for (int p = 0; p < 4; ++p) {
      int idx = t + p * NTHREADS;
      // A tile: 128 rows x 32 K
      int r  = idx >> 3;
      int kc = (idx & 7) << 2;
      ra[p] = *(const float4*)(Ag + (size_t)(mbase + r) * CDIM + k0 + kc);
      // B tile: 32 K-rows x 128 N (coalesced along N)
      int kr = idx >> 5;                   // 0..31
      int nc = (idx & 31) << 2;            // 0..124
      rb[p] = *(const float4*)(Dg + (size_t)(k0 + kr) * HWDIM + nbase + nc);
    }
  };
  auto lstore = [&](int buf) {
#pragma unroll
    for (int p = 0; p < 4; ++p) {
      int idx = t + p * NTHREADS;
      int r  = idx >> 3;
      int kc = (idx & 7) << 2;
      v4bf pa = { (bf16_t)ra[p].x, (bf16_t)ra[p].y, (bf16_t)ra[p].z, (bf16_t)ra[p].w };
      *(v4bf*)&sA[buf][r * LDSS + kc] = pa;
      // transpose depth into [n][k]
      int kr = idx >> 5;
      int nc = (idx & 31) << 2;
      sB[buf][(nc + 0) * LDSS + kr] = (bf16_t)rb[p].x;
      sB[buf][(nc + 1) * LDSS + kr] = (bf16_t)rb[p].y;
      sB[buf][(nc + 2) * LDSS + kr] = (bf16_t)rb[p].z;
      sB[buf][(nc + 3) * LDSS + kr] = (bf16_t)rb[p].w;
    }
  };

  v8f acc[4][2] = {};

  gload(0);
  lstore(0);
  __syncthreads();

  const int NK = CDIM / BK;                // 16 K-steps
  for (int kt = 0; kt < NK; ++kt) {
    const int cur = kt & 1;
    if (kt + 1 < NK) gload((kt + 1) * BK);

    v16bf af[4], bf[2];
#pragma unroll
    for (int i = 0; i < 4; ++i)
      af[i] = frag_a(&sA[cur][(wm * 64 + i * 16) * LDSS], lane);
#pragma unroll
    for (int j = 0; j < 2; ++j)
      bf[j] = frag_b(&sB[cur][(wn * 32 + j * 16) * LDSS], lane);

#pragma unroll
    for (int i = 0; i < 4; ++i)
#pragma unroll
      for (int j = 0; j < 2; ++j)
        acc[i][j] = __builtin_amdgcn_wmma_f32_16x16x32_bf16(
            false, af[i], false, bf[j], (short)0, acc[i][j], false, false);

    if (kt + 1 < NK) lstore(cur ^ 1);
    __syncthreads();
  }

  // epilogue: out = rgb + acc
  const int half = lane >> 4;
  const int ln   = lane & 15;
#pragma unroll
  for (int i = 0; i < 4; ++i) {
#pragma unroll
    for (int j = 0; j < 2; ++j) {
      const int n = nbase + wn * 32 + j * 16 + ln;
#pragma unroll
      for (int v = 0; v < 8; ++v) {
        const int m = mbase + wm * 64 + i * 16 + half * 8 + v;
        const size_t off = (size_t)m * HWDIM + n;
        Og[off] = Rg[off] + acc[i][j][v];
      }
    }
  }
}

extern "C" void kernel_launch(void* const* d_in, const int* in_sizes, int n_in,
                              void* d_out, int out_size, void* d_ws, size_t ws_size,
                              hipStream_t stream) {
  (void)in_sizes; (void)n_in; (void)out_size; (void)ws_size;
  const float* rgb   = (const float*)d_in[0];
  const float* depth = (const float*)d_in[1];
  float* out = (float*)d_out;
  float* S   = (float*)d_ws;               // needs 16*512*512*4 = 16 MiB

  dim3 blk(NTHREADS, 1, 1);
  dim3 g1(CDIM / BN, CDIM / BM, BATCH);    // (4, 4, 16)
  ff_gemm1_sigmoid<<<g1, blk, 0, stream>>>(rgb, depth, S);

  dim3 g2(HWDIM / BN, CDIM / BM, BATCH);   // (32, 4, 16)
  ff_gemm2_add<<<g2, blk, 0, stream>>>(rgb, depth, S, out);
}
